// CascadedGroupAttention_549755814644
// MI455X (gfx1250) — compile-verified, matrix operand order checked by reading, not run on hardware
//
#include <hip/hip_runtime.h>
#include <hip/hip_bf16.h>

typedef __attribute__((ext_vector_type(16))) _Float16 v16h;
typedef __attribute__((ext_vector_type(8)))  float    v8f;

#define RES_   7
#define N_     49
#define NP_    64      // padded row count for all LDS tiles
#define C_     512
#define HEADS_ 4
#define KD_    32
#define D_     128
#define KS_    5
#define QO_    192     // 2*KD + D
#define EPS_   1e-5f

#define FEAT_LD 128
#define ATTN_LD 64
#define H_LD    512

#define WMMA_F16(a, b, c) \
    __builtin_amdgcn_wmma_f32_16x16x32_f16(false, (a), false, (b), (short)0, (c), false, false)

// ---------------------------------------------------------------------------
// Single fragment loader: contiguous f16 rows in LDS (used for both A and,
// via the mirrored layout, B). 16-bit A 16x32 layout (ISA 7.12.2):
// lane L -> row = L%16, half = L/16; VGPR r holds K pair:
// k = (r<4 ? 2r : 16+2(r-4)) + 8*half  => k0..7 and k16..23 contiguous,
// lowering to 2x ds_load_b128 per fragment, no converts.
// ---------------------------------------------------------------------------
__device__ inline v16h load_frag_h(const _Float16* src, int ld, int m0, int k0) {
    int lane = threadIdx.x & 31;
    int half = lane >> 4;
    int m = m0 + (lane & 15);
    const _Float16* p = src + m * ld + k0 + 8 * half;
    v16h a;
#pragma unroll
    for (int r = 0; r < 8; ++r) {
        int k = (r < 4) ? (2 * r) : (16 + 2 * (r - 4));
        a[2 * r]     = p[k];
        a[2 * r + 1] = p[k + 1];
    }
    return a;
}

// B(k,n) = W[n*ld + k], f16 weights ([out][in] row-major) from global
__device__ inline v16h load_frag_b_wt(const _Float16* W, int ld, int n0, int k0) {
    int lane = threadIdx.x & 31;
    int half = lane >> 4;
    int n = n0 + (lane & 15);
    const _Float16* p = W + n * ld + k0 + 8 * half;
    v16h b;
#pragma unroll
    for (int r = 0; r < 8; ++r) {
        int k = (r < 4) ? (2 * r) : (16 + 2 * (r - 4));
        b[2 * r]     = p[k];
        b[2 * r + 1] = p[k + 1];
    }
    return b;
}

// ---------------------------------------------------------------------------
// Prep kernels
// ---------------------------------------------------------------------------
extern "C" __global__ void cga_prep_bias(const float* __restrict__ attn_bias,
                                         const int* __restrict__ bias_idxs,
                                         int noff, float* __restrict__ ab) {
    int idx = blockIdx.x * 256 + threadIdx.x;
    if (idx >= HEADS_ * N_ * N_) return;
    int h  = idx / (N_ * N_);
    int nm = idx % (N_ * N_);
    ab[idx] = attn_bias[h * noff + bias_idxs[nm]];
}

extern "C" __global__ void cga_prep_bn(const float* __restrict__ g,
                                       const float* __restrict__ bb,
                                       const float* __restrict__ m,
                                       const float* __restrict__ v,
                                       float* __restrict__ s,
                                       float* __restrict__ t, int n) {
    int i = blockIdx.x * 256 + threadIdx.x;
    if (i >= n) return;
    float sv = g[i] * rsqrtf(v[i] + EPS_);
    s[i] = sv;
    t[i] = bb[i] - m[i] * sv;
}

extern "C" __global__ void cga_f32_to_f16(const float* __restrict__ src,
                                          _Float16* __restrict__ dst, int n) {
    int i = blockIdx.x * 256 + threadIdx.x;
    if (i < n) dst[i] = (_Float16)src[i];
}

// ---------------------------------------------------------------------------
// Main kernel: one workgroup (8 wave32s) per batch element.
// All WMMA operands are kept as f16 in LDS (converted once at producers);
// waves own an M-tile and reuse the A fragment across an N-strip.
// ---------------------------------------------------------------------------
extern "C" __global__ __launch_bounds__(256)
void cga_main(const float* __restrict__ x,
              const float* __restrict__ dw_w,
              const float* __restrict__ ab,
              const float* __restrict__ qkv_s, const float* __restrict__ qkv_t,
              const float* __restrict__ dw_s,  const float* __restrict__ dw_t,
              const float* __restrict__ proj_s,const float* __restrict__ proj_t,
              const _Float16* __restrict__ qkv_wh,
              const _Float16* __restrict__ proj_wh,
              float* __restrict__ out) {
    extern __shared__ char smem[];
    float*     feat   = (float*)smem;                        // 64x128 f32 (cascade master)
    float*     qraw   = feat + NP_ * FEAT_LD;                // 64x32  f32 (pre-conv q)
    float*     atf    = qraw + NP_ * KD_;                    // 64x64  f32 (attn scores)
    _Float16*  feat16 = (_Float16*)(atf + NP_ * ATTN_LD);    // 64x128 f16
    _Float16*  kh16   = feat16 + NP_ * FEAT_LD;              // 64x32  f16 (k)
    _Float16*  vt16   = kh16 + NP_ * KD_;                    // 128x64 f16 (v transposed)
    _Float16*  qb16   = vt16 + D_ * NP_;                     // 64x32  f16 (conv out)
    _Float16*  at16   = qb16 + NP_ * KD_;                    // 64x64  f16 (softmax out)
    _Float16*  hb     = at16 + NP_ * ATTN_LD;                // 64x512 f16 (relu concat)

    const int b    = blockIdx.x;
    const int tid  = threadIdx.x;
    const int wave = tid >> 5;
    const int lane = tid & 31;
    const int lrow = lane & 15;      // row/col within tile
    const int lhalf = lane >> 4;
    const float scale = 0.17677669529663687f;  // KD^-0.5

    const float* xb = x + (size_t)b * N_ * C_;

    // One-time zeroing: feat16 pad rows (feed GEMM1 A) and ALL of at16
    // (AV A operand: softmax only ever writes rows/cols < 49).
    for (int idx = tid; idx < (NP_ - N_) * FEAT_LD; idx += 256)
        feat16[N_ * FEAT_LD + idx] = (_Float16)0.f;
    for (int idx = tid; idx < NP_ * ATTN_LD; idx += 256)
        at16[idx] = (_Float16)0.f;
    __syncthreads();

    for (int head = 0; head < HEADS_; ++head) {
        // feat (+)= chunks[head]; refresh f16 mirror in the same pass
        for (int idx = tid; idx < N_ * D_; idx += 256) {
            int n = idx / D_, c = idx % D_;
            float v = xb[n * C_ + head * D_ + c];
            float f = (head == 0) ? v : feat[n * FEAT_LD + c] + v;
            feat[n * FEAT_LD + c] = f;
            feat16[n * FEAT_LD + c] = (_Float16)f;
        }
        __syncthreads();

        // ---- GEMM1: y(64x192) = feat @ Wqkv^T, +BN; scatter to q/k/v^T ----
        {
            const _Float16* Wh = qkv_wh + head * QO_ * D_;
            const float*    s1 = qkv_s + head * QO_;
            const float*    t1 = qkv_t + head * QO_;
            const int mt = wave >> 1;
            const int ntbase = (wave & 1) * 6;     // 12 N-tiles split 6/6
            v8f acc[6] = {};
#pragma unroll
            for (int kk = 0; kk < 4; ++kk) {
                v16h a = load_frag_h(feat16, FEAT_LD, mt * 16, kk * 32);
#pragma unroll
                for (int j = 0; j < 6; ++j) {
                    v16h bf = load_frag_b_wt(Wh, D_, (ntbase + j) * 16, kk * 32);
                    acc[j] = WMMA_F16(a, bf, acc[j]);
                }
            }
#pragma unroll
            for (int j = 0; j < 6; ++j) {
                int ntile = ntbase + j;            // wave-uniform
                int n = ntile * 16 + lrow;
                float sv = s1[n], tv = t1[n];
                if (ntile < 2) {                   // q -> f32 for depthwise conv
#pragma unroll
                    for (int r = 0; r < 8; ++r) {
                        int m = mt * 16 + r + 8 * lhalf;
                        qraw[m * KD_ + n] = acc[j][r] * sv + tv;
                    }
                } else if (ntile < 4) {            // k -> f16
#pragma unroll
                    for (int r = 0; r < 8; ++r) {
                        int m = mt * 16 + r + 8 * lhalf;
                        kh16[m * KD_ + (n - KD_)] = (_Float16)(acc[j][r] * sv + tv);
                    }
                } else {                           // v -> f16 TRANSPOSED [d][m]
                    int d = n - 2 * KD_;
                    int mb = mt * 16 + 8 * lhalf;
#pragma unroll
                    for (int r = 0; r < 8; ++r)
                        vt16[d * NP_ + mb + r] = (_Float16)(acc[j][r] * sv + tv);
                }
            }
        }
        __syncthreads();

        // ---- Depthwise 5x5 conv on q (49x32) + BN -> qb16 -----------------
        {
            const float* wdw = dw_w + head * KD_ * KS_ * KS_;
            const float* s2  = dw_s + head * KD_;
            const float* t2  = dw_t + head * KD_;
            for (int idx = tid; idx < N_ * KD_; idx += 256) {
                int n = idx / KD_, c = idx % KD_;
                int py = n / RES_, px = n % RES_;
                float accv = 0.f;
#pragma unroll
                for (int ky = 0; ky < KS_; ++ky) {
                    int iy = py + ky - 2;
                    if (iy < 0 || iy >= RES_) continue;
#pragma unroll
                    for (int kx = 0; kx < KS_; ++kx) {
                        int ix = px + kx - 2;
                        if (ix < 0 || ix >= RES_) continue;
                        accv += qraw[(iy * RES_ + ix) * KD_ + c] *
                                wdw[c * KS_ * KS_ + ky * KS_ + kx];
                    }
                }
                qb16[n * KD_ + c] = (_Float16)(accv * s2[c] + t2[c]);
            }
        }
        __syncthreads();

        // ---- attn f32 = q_conv @ k^T * scale + bias -----------------------
        // Pads of atf are never read (softmax touches rows/cols < 49 only),
        // so stores are unconditional with a clamped bias index.
        {
            const float* abh = ab + head * N_ * N_;
            const int mt = wave >> 1;
            const int ntbase = (wave & 1) * 2;     // 4 N-tiles split 2/2
            v16h a = load_frag_h(qb16, KD_, mt * 16, 0);
            v8f acc[2] = {};
#pragma unroll
            for (int j = 0; j < 2; ++j) {
                v16h bf = load_frag_h(kh16, KD_, (ntbase + j) * 16, 0);
                acc[j] = WMMA_F16(a, bf, acc[j]);
            }
#pragma unroll
            for (int j = 0; j < 2; ++j) {
                int n  = (ntbase + j) * 16 + lrow;
                int bn = (n < N_) ? n : (N_ - 1);
#pragma unroll
                for (int r = 0; r < 8; ++r) {
                    int m  = mt * 16 + r + 8 * lhalf;
                    int bm = (m < N_) ? m : (N_ - 1);
                    atf[m * ATTN_LD + n] = acc[j][r] * scale + abh[bm * N_ + bn];
                }
            }
        }
        __syncthreads();

        // ---- softmax over keys; writes f16 result into at16 ---------------
        if (tid < N_) {
            float* row = atf + tid * ATTN_LD;
            _Float16* row16 = at16 + tid * ATTN_LD;
            float mx = -1e30f;
            for (int j = 0; j < N_; ++j) mx = fmaxf(mx, row[j]);
            float sum = 0.f;
            for (int j = 0; j < N_; ++j) {
                float e = __expf(row[j] - mx);
                row[j] = e;
                sum += e;
            }
            float inv = 1.f / sum;
            for (int j = 0; j < N_; ++j) row16[j] = (_Float16)(row[j] * inv);
        }
        __syncthreads();

        // ---- feat(64x128) = attn @ v; h = relu(feat) in f16 ---------------
        // at16 pads are exactly zero => padded K=64 is safe, pad-row outputs
        // are finite zeros.
        {
            const int mt = wave >> 1;
            const int ntbase = (wave & 1) * 4;     // 8 N-tiles split 4/4
            v8f acc[4] = {};
#pragma unroll
            for (int kk = 0; kk < 2; ++kk) {
                v16h a = load_frag_h(at16, ATTN_LD, mt * 16, kk * 32);
#pragma unroll
                for (int j = 0; j < 4; ++j) {
                    v16h bf = load_frag_h(vt16, NP_, (ntbase + j) * 16, kk * 32);
                    acc[j] = WMMA_F16(a, bf, acc[j]);
                }
            }
#pragma unroll
            for (int j = 0; j < 4; ++j) {
                int n = (ntbase + j) * 16 + lrow;
#pragma unroll
                for (int r = 0; r < 8; ++r) {
                    int m = mt * 16 + r + 8 * lhalf;
                    feat[m * FEAT_LD + n] = acc[j][r];
                    hb[m * H_LD + head * D_ + n] = (_Float16)fmaxf(acc[j][r], 0.f);
                }
            }
        }
        __syncthreads();
    }

    // ---- projection: out(49x512) = h @ proj_w^T, +BN fold -----------------
    {
        float* ob = out + (size_t)b * N_ * C_;
        const int mt = wave >> 1;
        const int ntbase = (wave & 1) * 16;        // 32 N-tiles split 16/16
#pragma unroll 1
        for (int nb = 0; nb < 4; ++nb) {
            v8f acc[4] = {};
            for (int kk = 0; kk < 16; ++kk) {
                v16h a = load_frag_h(hb, H_LD, mt * 16, kk * 32);
#pragma unroll
                for (int j = 0; j < 4; ++j) {
                    v16h bf = load_frag_b_wt(proj_wh, C_,
                                             (ntbase + nb * 4 + j) * 16, kk * 32);
                    acc[j] = WMMA_F16(a, bf, acc[j]);
                }
            }
#pragma unroll
            for (int j = 0; j < 4; ++j) {
                int n = (ntbase + nb * 4 + j) * 16 + lrow;
                float sv = proj_s[n], tv = proj_t[n];
#pragma unroll
                for (int r = 0; r < 8; ++r) {
                    int m = mt * 16 + r + 8 * lhalf;
                    if (m < N_) ob[m * C_ + n] = acc[j][r] * sv + tv;
                }
            }
        }
    }
}

// ---------------------------------------------------------------------------
// Launch
// ---------------------------------------------------------------------------
extern "C" void kernel_launch(void* const* d_in, const int* in_sizes, int n_in,
                              void* d_out, int out_size, void* d_ws, size_t ws_size,
                              hipStream_t stream) {
    const float* x        = (const float*)d_in[0];
    const float* qkv_w    = (const float*)d_in[1];
    const float* qkv_g    = (const float*)d_in[2];
    const float* qkv_b    = (const float*)d_in[3];
    const float* qkv_m    = (const float*)d_in[4];
    const float* qkv_v    = (const float*)d_in[5];
    const float* dw_w     = (const float*)d_in[6];
    const float* dw_g     = (const float*)d_in[7];
    const float* dw_b     = (const float*)d_in[8];
    const float* dw_m     = (const float*)d_in[9];
    const float* dw_v     = (const float*)d_in[10];
    const float* proj_w   = (const float*)d_in[11];
    const float* proj_g   = (const float*)d_in[12];
    const float* proj_b   = (const float*)d_in[13];
    const float* proj_m   = (const float*)d_in[14];
    const float* proj_v   = (const float*)d_in[15];
    const float* attn_bias= (const float*)d_in[16];
    const int*   bias_idxs= (const int*)d_in[17];
    float* out = (float*)d_out;

    const int Bn   = in_sizes[0] / (N_ * C_);
    const int noff = in_sizes[16] / HEADS_;

    char* ws = (char*)d_ws;
    size_t off = 0;
    auto carve = [&](size_t bytes) {
        void* p = ws + off;
        off = (off + bytes + 255) & ~(size_t)255;
        return p;
    };
    float*    ab      = (float*)carve(HEADS_ * N_ * N_ * sizeof(float));
    float*    qkv_s   = (float*)carve(HEADS_ * QO_ * sizeof(float));
    float*    qkv_t   = (float*)carve(HEADS_ * QO_ * sizeof(float));
    float*    dw_s    = (float*)carve(HEADS_ * KD_ * sizeof(float));
    float*    dw_t    = (float*)carve(HEADS_ * KD_ * sizeof(float));
    float*    proj_s  = (float*)carve(C_ * sizeof(float));
    float*    proj_t  = (float*)carve(C_ * sizeof(float));
    _Float16* qkv_wh  = (_Float16*)carve(HEADS_ * QO_ * D_ * sizeof(_Float16));
    _Float16* proj_wh = (_Float16*)carve(C_ * C_ * sizeof(_Float16));

    {
        int n = HEADS_ * N_ * N_;
        cga_prep_bias<<<(n + 255) / 256, 256, 0, stream>>>(attn_bias, bias_idxs, noff, ab);
    }
    cga_prep_bn<<<(HEADS_ * QO_ + 255) / 256, 256, 0, stream>>>(
        qkv_g, qkv_b, qkv_m, qkv_v, qkv_s, qkv_t, HEADS_ * QO_);
    cga_prep_bn<<<(HEADS_ * KD_ + 255) / 256, 256, 0, stream>>>(
        dw_g, dw_b, dw_m, dw_v, dw_s, dw_t, HEADS_ * KD_);
    cga_prep_bn<<<(C_ + 255) / 256, 256, 0, stream>>>(
        proj_g, proj_b, proj_m, proj_v, proj_s, proj_t, C_);
    cga_f32_to_f16<<<(HEADS_ * QO_ * D_ + 255) / 256, 256, 0, stream>>>(
        qkv_w, qkv_wh, HEADS_ * QO_ * D_);
    cga_f32_to_f16<<<(C_ * C_ + 255) / 256, 256, 0, stream>>>(
        proj_w, proj_wh, C_ * C_);

    // LDS: feat(32K) + qraw(8K) + atf(16K) + feat16(16K) + kh16(4K)
    //    + vt16(16K) + qb16(4K) + at16(8K) + hb(64K) = 168 KB (< 320 KB/WGP)
    constexpr size_t SMEM =
        (size_t)NP_ * FEAT_LD * 4 +      // feat f32
        (size_t)NP_ * KD_     * 4 +      // qraw f32
        (size_t)NP_ * ATTN_LD * 4 +      // atf f32
        (size_t)NP_ * FEAT_LD * 2 +      // feat16
        (size_t)NP_ * KD_     * 2 +      // kh16
        (size_t)D_  * NP_     * 2 +      // vt16
        (size_t)NP_ * KD_     * 2 +      // qb16
        (size_t)NP_ * ATTN_LD * 2 +      // at16
        (size_t)NP_ * H_LD    * 2;       // hb
    cga_main<<<dim3(Bn), dim3(256), SMEM, stream>>>(
        x, dw_w, ab, qkv_s, qkv_t, dw_s, dw_t, proj_s, proj_t,
        qkv_wh, proj_wh, out);
}
